// Mamba2DStable_24361054503033
// MI455X (gfx1250) — matrix operand order repaired and checked by reading, not compiled, not run on hardware
//
#include <hip/hip_runtime.h>
#include <hip/hip_bf16.h>

// ---- shapes (compile-time) ----
#define BATCH   2
#define DIM     512
#define D_STATE 16
#define D_CONV  4
#define D_INNER 1024          // EXPAND * DIM
#define DT_RANK 32
#define LLEN    4096          // H*W = 64*64
#define MROWS   (BATCH*LLEN)  // 8192

typedef _Float16 v16h __attribute__((ext_vector_type(16)));
typedef _Float16 v8h  __attribute__((ext_vector_type(8)));
typedef float    v8f  __attribute__((ext_vector_type(8)));

__device__ __forceinline__ float clamp10(float v) {
    return fminf(10.f, fmaxf(-10.f, v));
}
__device__ __forceinline__ float silu(float v) {
    return v / (1.f + __expf(-v));
}

// ---------------- f32 -> f16 convert ----------------
__global__ void k_f32_to_f16(const float* __restrict__ src, _Float16* __restrict__ dst, int n) {
    int i = blockIdx.x * blockDim.x + threadIdx.x;
    if (i < n) dst[i] = (_Float16)src[i];
}

// ---------------- clamp + LayerNorm over C, gather (B,C,L)->(B,L,C) f16 ----------------
__global__ void k_ln_in(const float* __restrict__ x, const float* __restrict__ w,
                        const float* __restrict__ b, _Float16* __restrict__ xs16) {
    int row  = blockIdx.x;          // row = bidx*LLEN + l
    int bidx = row >> 12;           // / LLEN
    int l    = row & (LLEN - 1);
    int tid  = threadIdx.x;         // 256 threads, 2 elems each (C=512)
    const float* xr = x + (size_t)bidx * DIM * LLEN + l;   // x[b][c][l], stride LLEN per c

    float v0 = clamp10(xr[(size_t)tid * LLEN]);
    float v1 = clamp10(xr[(size_t)(tid + 256) * LLEN]);

    __shared__ float s1[256], s2[256];
    s1[tid] = v0 + v1;
    s2[tid] = v0 * v0 + v1 * v1;
    __syncthreads();
    for (int off = 128; off > 0; off >>= 1) {
        if (tid < off) { s1[tid] += s1[tid + off]; s2[tid] += s2[tid + off]; }
        __syncthreads();
    }
    float mu  = s1[0] * (1.f / DIM);
    float var = s2[0] * (1.f / DIM) - mu * mu;
    float rs  = rsqrtf(var + 1e-5f);

    _Float16* orow = xs16 + (size_t)row * DIM;
    orow[tid]       = (_Float16)(((v0 - mu) * rs) * w[tid] + b[tid]);
    orow[tid + 256] = (_Float16)(((v1 - mu) * rs) * w[tid + 256] + b[tid + 256]);
}

// ---------------- WMMA GEMM: out[M,N] = A[M,K] * W[N,K]^T ----------------
// A tile (16 rows x K) staged in LDS, shared by all waves in the block.
// Each wave computes a 16 x (16*NT) strip: one A fragment feeds NT WMMAs.
// K is a template constant (multiple of 32); tiling is exact (EXEC all ones).
// epi==1: += bias[n], softplus.
template <int K, int NT>
__global__ void k_wmma_gemm(const _Float16* __restrict__ A, const _Float16* __restrict__ Wt,
                            float* __restrict__ out, int ldc,
                            const float* __restrict__ bias, int epi) {
    const int lane    = threadIdx.x & 31;
    const int wave    = threadIdx.x >> 5;
    const int m_base  = blockIdx.x << 4;
    const int n_base0 = (blockIdx.y * (blockDim.x >> 5) + wave) * (16 * NT);

    // stage the 16 x K A tile into LDS (rows are contiguous, lda == K)
    __shared__ alignas(16) _Float16 As[16 * K];
    {
        const uint4* src = (const uint4*)(A + (size_t)m_base * K);
        uint4* dst = (uint4*)As;
        for (int i = threadIdx.x; i < (16 * K) / 8; i += blockDim.x) dst[i] = src[i];
    }
    __syncthreads();

    const int am = lane & 15, ag = lane >> 4;   // A: row m, K-half-group g
    const int bn = lane & 15, bg = lane >> 4;   // B: col n, K-half-group g

    v8f acc[NT];
    #pragma unroll
    for (int j = 0; j < NT; ++j) acc[j] = (v8f){};

    const _Float16* asrc = As + am * K;
    #pragma unroll 1
    for (int k0 = 0; k0 < K; k0 += 32) {
        // A 16x32 f16 layout: lanes g=0 -> K {0..7,16..23}, g=1 -> K {8..15,24..31}
        v8h alo = *(const v8h*)(asrc + k0 + ag * 8);
        v8h ahi = *(const v8h*)(asrc + k0 + 16 + ag * 8);
        v16h av = __builtin_shufflevector(alo, ahi, 0,1,2,3,4,5,6,7,8,9,10,11,12,13,14,15);
        #pragma unroll
        for (int j = 0; j < NT; ++j) {
            // B 32x16 f16 layout: lanes g=0 -> K {0..7,16..23}, g=1 -> K {8..15,24..31} (ldb == K)
            const _Float16* brow = Wt + (size_t)(n_base0 + j * 16 + bn) * K;
            v8h blo = *(const v8h*)(brow + k0 + bg * 8);
            v8h bhi = *(const v8h*)(brow + k0 + 16 + bg * 8);
            v16h bv = __builtin_shufflevector(blo, bhi, 0,1,2,3,4,5,6,7,8,9,10,11,12,13,14,15);
            acc[j] = __builtin_amdgcn_wmma_f32_16x16x32_f16(false, av, false, bv,
                                                            (short)0, acc[j], false, false);
        }
    }

    // D f32 16x16: acc[j][v] = D[g*8+v][n]
    const int n = lane & 15;
    const int g = lane >> 4;
    #pragma unroll
    for (int j = 0; j < NT; ++j) {
        int col = n_base0 + j * 16 + n;
        #pragma unroll
        for (int v = 0; v < 8; ++v) {
            int row = m_base + g * 8 + v;
            float val = acc[j][v];
            if (epi == 1) {
                val += bias[col];
                val = (val > 20.f) ? val : log1pf(__expf(val));   // softplus
            }
            out[(size_t)row * ldc + col] = val;
        }
    }
}

// ---------------- depthwise causal conv (k=4) along L + bias + silu ----------------
__global__ void k_conv(const float* __restrict__ xz, const float* __restrict__ cw,
                       const float* __restrict__ cb, float* __restrict__ xc,
                       _Float16* __restrict__ xc16) {
    size_t idx = (size_t)blockIdx.x * blockDim.x + threadIdx.x;  // over B*L*D_INNER
    int d   = idx & (D_INNER - 1);
    size_t r = idx >> 10;            // row = b*LLEN + l
    int l   = (int)(r & (LLEN - 1));
    float acc = cb[d];
    #pragma unroll
    for (int j = 0; j < D_CONV; ++j) {
        int ll = l - (D_CONV - 1) + j;
        if (ll >= 0)
            acc += cw[d * D_CONV + j] * xz[(r + (ll - l)) * (2 * D_INNER) + d];
    }
    float s = silu(acc);
    xc[idx]   = s;
    xc16[idx] = (_Float16)s;
}

// ---------------- extract dt columns of xdbl -> f16 ----------------
__global__ void k_dt_extract(const float* __restrict__ xdbl, _Float16* __restrict__ dt16) {
    size_t idx = (size_t)blockIdx.x * blockDim.x + threadIdx.x;  // over MROWS*DT_RANK
    size_t row = idx >> 5;           // / 32
    int    c   = (int)(idx & 31);
    dt16[idx] = (_Float16)xdbl[row * (DT_RANK + 2 * D_STATE) + c];
}

// ---------------- selective scan ----------------
// lane = (channel-pair, state): 16 lanes per channel, h scalar per lane.
// per step: 1 exp + 2 fma + 4-step shfl_xor reduction over the 16 state lanes.
__global__ void k_scan(const float* __restrict__ dt, const float* __restrict__ xdbl,
                       const float* __restrict__ xc, const float* __restrict__ A_log,
                       float* __restrict__ y) {
    const int lane = threadIdx.x & 31;
    const int n    = lane & 15;           // state index
    const int half = lane >> 4;           // which channel of the pair
    const int wave = threadIdx.x >> 5;
    const int d = blockIdx.x * ((int)blockDim.x >> 4) + wave * 2 + half;
    const int b = blockIdx.y;

    const float Aneg = -__expf(A_log[d * D_STATE + n]);
    float h = 0.f;
    const size_t base = (size_t)b * LLEN;
    for (int t = 0; t < LLEN; ++t) {
        size_t r = base + t;
        float dtv = dt[r * D_INNER + d];
        float xv  = xc[r * D_INNER + d];
        const float* bc = xdbl + r * (DT_RANK + 2 * D_STATE);
        float Bt = bc[DT_RANK + n];
        float Ct = bc[DT_RANK + D_STATE + n];
        h = __expf(dtv * Aneg) * h + (dtv * xv) * Bt;
        float p = h * Ct;
        #pragma unroll
        for (int m = 8; m > 0; m >>= 1) p += __shfl_xor(p, m, 16);
        if (n == 0) y[r * D_INNER + d] = p;
    }
}

// ---------------- y = (y + xc*D) * silu(z) -> f16 ----------------
__global__ void k_ycombine(const float* __restrict__ y, const float* __restrict__ xc,
                           const float* __restrict__ Dp, const float* __restrict__ xz,
                           _Float16* __restrict__ y16) {
    size_t idx = (size_t)blockIdx.x * blockDim.x + threadIdx.x;  // over MROWS*D_INNER
    int d = idx & (D_INNER - 1);
    size_t row = idx >> 10;
    float zv = xz[row * (2 * D_INNER) + D_INNER + d];
    float v  = (y[idx] + xc[idx] * Dp[d]) * silu(zv);
    y16[idx] = (_Float16)v;
}

// ---------------- clamp + LayerNorm (w=1,b=0) + clamp, scatter to (B,C,H,W) ----------------
__global__ void k_ln_out(const float* __restrict__ opre, float* __restrict__ out) {
    int row  = blockIdx.x;
    int bidx = row >> 12;
    int l    = row & (LLEN - 1);
    int tid  = threadIdx.x;
    const float* xr = opre + (size_t)row * DIM;
    float v0 = clamp10(xr[tid]);
    float v1 = clamp10(xr[tid + 256]);

    __shared__ float s1[256], s2[256];
    s1[tid] = v0 + v1;
    s2[tid] = v0 * v0 + v1 * v1;
    __syncthreads();
    for (int off = 128; off > 0; off >>= 1) {
        if (tid < off) { s1[tid] += s1[tid + off]; s2[tid] += s2[tid + off]; }
        __syncthreads();
    }
    float mu  = s1[0] * (1.f / DIM);
    float var = s2[0] * (1.f / DIM) - mu * mu;
    float rs  = rsqrtf(var + 1e-5f);

    float* ob = out + (size_t)bidx * DIM * LLEN + l;
    ob[(size_t)tid * LLEN]         = clamp10((v0 - mu) * rs);
    ob[(size_t)(tid + 256) * LLEN] = clamp10((v1 - mu) * rs);
}

// ================= host launcher =================
static inline size_t align256(size_t x) { return (x + 255) & ~(size_t)255; }

extern "C" void kernel_launch(void* const* d_in, const int* in_sizes, int n_in,
                              void* d_out, int out_size, void* d_ws, size_t ws_size,
                              hipStream_t stream) {
    const float* x          = (const float*)d_in[0];
    const float* norm_w     = (const float*)d_in[1];
    const float* norm_b     = (const float*)d_in[2];
    const float* in_proj_w  = (const float*)d_in[3];   // (2048, 512)
    const float* conv_w     = (const float*)d_in[4];   // (1024, 1, 4)
    const float* conv_b     = (const float*)d_in[5];
    const float* x_proj_w   = (const float*)d_in[6];   // (64, 1024)
    const float* dt_proj_w  = (const float*)d_in[7];   // (1024, 32)
    const float* dt_proj_b  = (const float*)d_in[8];
    const float* A_log      = (const float*)d_in[9];   // (1024, 16)
    const float* Dvec       = (const float*)d_in[10];
    const float* out_proj_w = (const float*)d_in[11];  // (512, 1024)
    float* out = (float*)d_out;

    // workspace layout
    char* ws = (char*)d_ws;
    size_t off = 0;
    auto take = [&](size_t bytes) { char* p = ws + off; off = align256(off + bytes); return p; };
    _Float16* xs16   = (_Float16*)take((size_t)MROWS * DIM * 2);
    _Float16* wip16  = (_Float16*)take((size_t)2 * D_INNER * DIM * 2);
    _Float16* wxp16  = (_Float16*)take((size_t)(DT_RANK + 2 * D_STATE) * D_INNER * 2);
    _Float16* wdt16  = (_Float16*)take((size_t)D_INNER * DT_RANK * 2);
    _Float16* wop16  = (_Float16*)take((size_t)DIM * D_INNER * 2);
    float*    xz     = (float*)take((size_t)MROWS * 2 * D_INNER * 4);
    float*    xc     = (float*)take((size_t)MROWS * D_INNER * 4);
    _Float16* xc16   = (_Float16*)take((size_t)MROWS * D_INNER * 2);
    float*    xdbl   = (float*)take((size_t)MROWS * (DT_RANK + 2 * D_STATE) * 4);
    _Float16* dt16   = (_Float16*)take((size_t)MROWS * DT_RANK * 2);
    float*    dtbuf  = (float*)take((size_t)MROWS * D_INNER * 4);
    float*    ybuf   = (float*)take((size_t)MROWS * D_INNER * 4);
    _Float16* y16    = (_Float16*)take((size_t)MROWS * D_INNER * 2);
    float*    opre   = (float*)take((size_t)MROWS * DIM * 4);
    (void)ws_size; (void)n_in; (void)in_sizes; (void)out_size;

    // 0) weight converts to f16
    {
        int n;
        n = 2 * D_INNER * DIM;
        k_f32_to_f16<<<(n + 255) / 256, 256, 0, stream>>>(in_proj_w, wip16, n);
        n = (DT_RANK + 2 * D_STATE) * D_INNER;
        k_f32_to_f16<<<(n + 255) / 256, 256, 0, stream>>>(x_proj_w, wxp16, n);
        n = D_INNER * DT_RANK;
        k_f32_to_f16<<<(n + 255) / 256, 256, 0, stream>>>(dt_proj_w, wdt16, n);
        n = DIM * D_INNER;
        k_f32_to_f16<<<(n + 255) / 256, 256, 0, stream>>>(out_proj_w, wop16, n);
    }

    // 1) clamp + layernorm -> xs16 (M, 512)
    k_ln_in<<<MROWS, 256, 0, stream>>>(x, norm_w, norm_b, xs16);

    // 2) in_proj: xz (M, 2048) = xs16 (M,512) @ wip16(2048,512)^T
    //    8 waves * 64 cols = 512 cols per block -> grid.y = 4
    k_wmma_gemm<DIM, 4><<<dim3(MROWS / 16, 4), 256, 0, stream>>>(
        xs16, wip16, xz, 2 * D_INNER, nullptr, 0);

    // 3) depthwise causal conv + silu -> xc, xc16
    {
        size_t n = (size_t)MROWS * D_INNER;
        k_conv<<<(unsigned)(n / 256), 256, 0, stream>>>(xz, conv_w, conv_b, xc, xc16);
    }

    // 4) x_proj: xdbl (M, 64) = xc16 (M,1024) @ wxp16(64,1024)^T  (1 wave, NT=4 covers N=64)
    k_wmma_gemm<D_INNER, 4><<<dim3(MROWS / 16, 1), 32, 0, stream>>>(
        xc16, wxp16, xdbl, DT_RANK + 2 * D_STATE, nullptr, 0);

    // 5) dt16 = f16(xdbl[:, :32])
    {
        size_t n = (size_t)MROWS * DT_RANK;
        k_dt_extract<<<(unsigned)(n / 256), 256, 0, stream>>>(xdbl, dt16);
    }

    // 6) dt GEMM + bias + softplus: dtbuf (M, 1024) = dt16 (M,32) @ wdt16(1024,32)^T
    //    8 waves * 64 cols = 512 per block -> grid.y = 2
    k_wmma_gemm<DT_RANK, 4><<<dim3(MROWS / 16, 2), 256, 0, stream>>>(
        dt16, wdt16, dtbuf, D_INNER, dt_proj_b, 1);

    // 7) selective scan: 16 state-lanes per channel, 2 channels per wave
    k_scan<<<dim3(D_INNER / 16, BATCH), 256, 0, stream>>>(dtbuf, xdbl, xc, A_log, ybuf);

    // 8) y = (y + xc*D) * silu(z) -> y16
    {
        size_t n = (size_t)MROWS * D_INNER;
        k_ycombine<<<(unsigned)(n / 256), 256, 0, stream>>>(ybuf, xc, Dvec, xz, y16);
    }

    // 9) out_proj: opre (M, 512) = y16 (M,1024) @ wop16(512,1024)^T  (8 waves * 64 = 512)
    k_wmma_gemm<D_INNER, 4><<<dim3(MROWS / 16, 1), 256, 0, stream>>>(
        y16, wop16, opre, DIM, nullptr, 0);

    // 10) clamp + LN + clamp, scatter to (B, C, H, W)
    k_ln_out<<<MROWS, 256, 0, stream>>>(opre, out);
}
